// HGNet_85495618995014
// MI455X (gfx1250) — compile-verified, matrix operand order checked by reading, not run on hardware
//
#include <hip/hip_runtime.h>
#include <hip/hip_bf16.h>

// ---------------------------------------------------------------------------
// HGNet point-cloud network for MI455X (gfx1250), wave32 + WMMA.
// All dense layers -> v_wmma_f32_16x16x32_f16 (f16 in, f32 accumulate).
// ---------------------------------------------------------------------------

typedef __attribute__((ext_vector_type(16))) _Float16 v16h;
typedef __attribute__((ext_vector_type(8)))  _Float16 v8h;
typedef __attribute__((ext_vector_type(8)))  float    v8f;

// ---------------- problem constants ----------------
#define BB   8
#define N0   2048
#define KNN  16
#define M1   512
#define M2   256

static const size_t E1  = (size_t)BB * N0 * KNN;   // 262144 edges, g1
static const size_t EP1 = (size_t)BB * M1 * KNN;   // 65536  edges, pool1 / g2
static const size_t EP2 = (size_t)BB * M2 * KNN;   // 32768  edges, pool2 / g3

// ---------------- workspace layout (in 4-byte words) ----------------
static const size_t O_IDX1  = 0;                         // int, E1
static const size_t O_X1    = O_IDX1  + E1;              // B*N0*64
static const size_t O_FPS1  = O_X1    + (size_t)BB*N0*64;// int, B*M1
static const size_t O_POS1S = O_FPS1  + (size_t)BB*M1;
static const size_t O_X1S   = O_POS1S + (size_t)BB*M1*3;
static const size_t O_NBR1  = O_X1S   + (size_t)BB*M1*64;// int, EP1
static const size_t O_NP1   = O_NBR1  + EP1;
static const size_t O_NF1   = O_NP1   + (size_t)BB*M1*3;
static const size_t O_IDX2  = O_NF1   + (size_t)BB*M1*64;// int, EP1
static const size_t O_X2    = O_IDX2  + EP1;             // B*M1*128
static const size_t O_FPS2  = O_X2    + (size_t)BB*M1*128;// int, B*M2
static const size_t O_POS2S = O_FPS2  + (size_t)BB*M2;
static const size_t O_X2S   = O_POS2S + (size_t)BB*M2*3;
static const size_t O_NBR2  = O_X2S   + (size_t)BB*M2*128;// int, EP2
static const size_t O_NP2   = O_NBR2  + EP2;
static const size_t O_NF2   = O_NP2   + (size_t)BB*M2*3;
static const size_t O_IDX3  = O_NF2   + (size_t)BB*M2*128;// int, EP2
static const size_t O_X3    = O_IDX3  + EP2;             // B*M2*256
static const size_t O_GF    = O_X3    + (size_t)BB*M2*256;
static const size_t O_FC1   = O_GF    + (size_t)BB*512;
static const size_t O_FC2   = O_FC1   + (size_t)BB*512;
static const size_t O_FC3   = O_FC2   + (size_t)BB*512;
static const size_t O_STATS = O_FC3   + (size_t)BB*384;  // 1024 words
static const size_t O_ARENA = O_STATS + 1024;

// arena sub-layouts (stages are sequential, arena is reused)
static const size_t A_HPRE1 = O_ARENA;                 // E1*32
static const size_t A_EF1   = A_HPRE1 + E1*32;         // E1*6
static const size_t A_MSG1  = A_HPRE1 + E1*32;         // E1*64 (overlaps EF1; EF1 dead)

static const size_t A_PD1 = O_ARENA;                   // EP1*3
static const size_t A_DH1 = A_PD1 + EP1*3;             // EP1*16
static const size_t A_DL1 = A_DH1 + EP1*16;            // EP1*64
static const size_t A_H1  = A_DL1 + EP1*64;            // EP1*64
static const size_t A_AH1 = A_H1  + EP1*64;            // EP1*16
static const size_t A_PW1 = A_AH1 + EP1*16;            // EP1*3
static const size_t A_FW1 = A_PW1 + EP1*3;             // EP1*64

static const size_t A_HP2  = O_ARENA;                  // EP1*64
static const size_t A_EF2  = A_HP2 + EP1*64;           // EP1*128
static const size_t A_MSG2 = A_EF2;                    // EP1*128 (overlap)

static const size_t A_PD2 = O_ARENA;                   // EP2*3
static const size_t A_DH2 = A_PD2 + EP2*3;             // EP2*16
static const size_t A_DL2 = A_DH2 + EP2*16;            // EP2*128
static const size_t A_H2  = A_DL2 + EP2*128;           // EP2*128
static const size_t A_AH2 = A_H2  + EP2*128;           // EP2*16
static const size_t A_PW2 = A_AH2 + EP2*16;            // EP2*3
static const size_t A_FW2 = A_PW2 + EP2*3;             // EP2*128

static const size_t A_HP3  = O_ARENA;                  // EP2*128
static const size_t A_EF3  = A_HP3 + EP2*128;          // EP2*256
static const size_t A_MSG3 = A_EF3;                    // EP2*256 (overlap)

// ---------------------------------------------------------------------------
// Kernels
// ---------------------------------------------------------------------------

__global__ void k_zero(float* p, size_t n) {
  size_t i = (size_t)blockIdx.x * blockDim.x + threadIdx.x;
  size_t st = (size_t)gridDim.x * blockDim.x;
  for (; i < n; i += st) p[i] = 0.0f;
}

// Brute-force kNN (k=16, smallest d2 first, tie -> lower index; self included)
template <int D>
__global__ void k_knn(const float* __restrict__ q, const float* __restrict__ r,
                      int B, int Mq, int Nr, int* __restrict__ out) {
  size_t tid = (size_t)blockIdx.x * blockDim.x + threadIdx.x;
  size_t tot = (size_t)B * Mq;
  if (tid >= tot) return;
  int b = (int)(tid / Mq);
  const float* qp = q + tid * (size_t)D;
  const float* rp = r + (size_t)b * Nr * D;
  float best[KNN]; int bidx[KNN];
  for (int i = 0; i < KNN; ++i) { best[i] = 3.0e38f; bidx[i] = 0; }
  for (int j = 0; j < Nr; ++j) {
    const float* rr = rp + (size_t)j * D;
    float d = 0.0f;
#pragma unroll
    for (int c = 0; c < D; ++c) { float t = qp[c] - rr[c]; d += t * t; }
    if (d < best[KNN - 1]) {
      int p = KNN - 1;
      while (p > 0 && best[p - 1] > d) { best[p] = best[p - 1]; bidx[p] = bidx[p - 1]; --p; }
      best[p] = d; bidx[p] = j;
    }
  }
  int* o = out + tid * KNN;
  for (int i = 0; i < KNN; ++i) o[i] = bidx[i];
}

// Farthest point sampling: one block (256 threads) per batch.
__global__ void k_fps(const float* __restrict__ pos, int N, int M, int* __restrict__ out) {
  const int T = 256;
  int b = blockIdx.x, t = threadIdx.x;
  const float* p = pos + (size_t)b * N * 3;
  int* o = out + (size_t)b * M;
  __shared__ float sval[T];
  __shared__ int   sidx[T];
  __shared__ int   scur;
  float mind[8];
  int npt = (N + T - 1) / T;
  for (int i = 0; i < 8; ++i) mind[i] = 1.0e10f;
  if (t == 0) { o[0] = 0; scur = 0; }
  __syncthreads();
  for (int it = 1; it < M; ++it) {
    int cur = scur;
    float cx = p[cur * 3], cy = p[cur * 3 + 1], cz = p[cur * 3 + 2];
    float bv = -1.0f; int bi = N;
    for (int i = 0; i < npt; ++i) {
      int n = t + i * T;
      if (n < N) {
        float dx = p[n * 3] - cx, dy = p[n * 3 + 1] - cy, dz = p[n * 3 + 2] - cz;
        float d = dx * dx + dy * dy + dz * dz;
        if (d < mind[i]) mind[i] = d;
        if (mind[i] > bv || (mind[i] == bv && n < bi)) { bv = mind[i]; bi = n; }
      }
    }
    sval[t] = bv; sidx[t] = bi;
    __syncthreads();
    for (int s = T / 2; s > 0; s >>= 1) {
      if (t < s) {
        if (sval[t + s] > sval[t] || (sval[t + s] == sval[t] && sidx[t + s] < sidx[t])) {
          sval[t] = sval[t + s]; sidx[t] = sidx[t + s];
        }
      }
      __syncthreads();
    }
    if (t == 0) { o[it] = sidx[0]; scur = sidx[0]; }
    __syncthreads();
  }
}

// dst[b,m,:] = src[b, idx[b*M+m], :]
__global__ void k_gather(const float* __restrict__ src, const int* __restrict__ idx,
                         float* __restrict__ dst, int B, int N, int M, int C) {
  size_t tot = (size_t)B * M * C;
  size_t i = (size_t)blockIdx.x * blockDim.x + threadIdx.x;
  if (i >= tot) return;
  size_t bm = i / C; int c = (int)(i % C);
  int b = (int)(bm / M);
  int j = idx[bm];
  dst[i] = src[((size_t)b * N + j) * C + c];
}

// ef[e] = concat(x_i, x_j - x_i), e = (b*N+n)*16+k
__global__ void k_edgefeat(const float* __restrict__ x, const int* __restrict__ kn,
                           float* __restrict__ ef, int B, int N, int C) {
  size_t E = (size_t)B * N * KNN;
  size_t e = (size_t)blockIdx.x * blockDim.x + threadIdx.x;
  if (e >= E) return;
  size_t bn = e >> 4;
  int b = (int)(bn / N), n = (int)(bn % N);
  int j = kn[e];
  const float* xi = x + ((size_t)b * N + n) * C;
  const float* xj = x + ((size_t)b * N + j) * C;
  float* o = ef + e * (size_t)(2 * C);
  for (int c = 0; c < C; ++c) { float a = xi[c]; o[c] = a; o[C + c] = xj[c] - a; }
}

// ---------------------------------------------------------------------------
// WMMA GEMM helpers
// ---------------------------------------------------------------------------
#define LWS 40   // LDS stride in halves per column (16B aligned, bank-spread)

__device__ __forceinline__ v16h gemm_load_a(const float* __restrict__ arow,
                                            int k0, int khalf, int Kd,
                                            bool mvalid, bool fastA) {
  v16h av;
  if (fastA) {
    const float* ap = arow + k0 + khalf;
    float4 a0 = *(const float4*)(ap);
    float4 a1 = *(const float4*)(ap + 4);
    float4 a2 = *(const float4*)(ap + 16);
    float4 a3 = *(const float4*)(ap + 20);
    av[0]  = (_Float16)a0.x; av[1]  = (_Float16)a0.y;
    av[2]  = (_Float16)a0.z; av[3]  = (_Float16)a0.w;
    av[4]  = (_Float16)a1.x; av[5]  = (_Float16)a1.y;
    av[6]  = (_Float16)a1.z; av[7]  = (_Float16)a1.w;
    av[8]  = (_Float16)a2.x; av[9]  = (_Float16)a2.y;
    av[10] = (_Float16)a2.z; av[11] = (_Float16)a2.w;
    av[12] = (_Float16)a3.x; av[13] = (_Float16)a3.y;
    av[14] = (_Float16)a3.z; av[15] = (_Float16)a3.w;
  } else {
#pragma unroll
    for (int j = 0; j < 8; ++j) {
      int kk = k0 + khalf + j;
      bool i0 = mvalid && (kk < Kd);
      float v0 = arow[i0 ? kk : 0];
      av[j] = (_Float16)(i0 ? v0 : 0.0f);
      int kk2 = kk + 16;
      bool i1 = mvalid && (kk2 < Kd);
      float v1 = arow[i1 ? kk2 : 0];
      av[8 + j] = (_Float16)(i1 ? v1 : 0.0f);
    }
  }
  return av;
}

__device__ __forceinline__ v16h gemm_load_b(const _Float16* __restrict__ lp, int khalf) {
  v8h blo = *(const v8h*)(lp + khalf);
  v8h bhi = *(const v8h*)(lp + khalf + 16);
  return __builtin_shufflevector(blo, bhi,
         0, 1, 2, 3, 4, 5, 6, 7, 8, 9, 10, 11, 12, 13, 14, 15);
}

// Generic WMMA GEMM (handles any N; used for N=16 and N=3 layers).
__global__ void __launch_bounds__(128) k_gemm(
    const float* __restrict__ A, const float* __restrict__ Wt,
    const float* __restrict__ bias, float* __restrict__ Cm,
    int M, int Kd, int N) {
  __shared__ _Float16 lwh[16 * LWS];
  const int lane  = threadIdx.x;
  const int wv    = threadIdx.y;
  const int tid   = wv * 32 + lane;
  const int tileM = blockIdx.x * 4 + wv;
  const int tileN = blockIdx.y;
  const int mrow  = lane & 15;
  const int nlo   = lane & 15;
  const int khalf = (lane >> 4) << 3;   // 0 or 8
  const int n0    = tileN * 16;
  const int n     = n0 + nlo;
  const int m     = tileM * 16 + mrow;
  const bool mvalid = (m < M);
  const bool fullM  = (tileM * 16 + 16 <= M);
  const bool fullN  = (n0 + 16 <= N);
  const float* arow = A + (size_t)(mvalid ? m : 0) * Kd;

  const int se    = tid * 4;
  const int skrow = se >> 4;
  const int sc0   = se & 15;

  v8f acc = {};
  for (int k0 = 0; k0 < Kd; k0 += 32) {
    const bool fullK = (k0 + 32 <= Kd);
    if (fullM && (k0 + 64 <= Kd))
      __builtin_prefetch(arow + k0 + 32, 0, 1);

    __syncthreads();
    {
      int kr = k0 + skrow;
      if (fullN && fullK) {
        const float4 w = *(const float4*)&Wt[(size_t)kr * N + n0 + sc0];
        lwh[(sc0 + 0) * LWS + skrow] = (_Float16)w.x;
        lwh[(sc0 + 1) * LWS + skrow] = (_Float16)w.y;
        lwh[(sc0 + 2) * LWS + skrow] = (_Float16)w.z;
        lwh[(sc0 + 3) * LWS + skrow] = (_Float16)w.w;
      } else {
#pragma unroll
        for (int u = 0; u < 4; ++u) {
          int cc = n0 + sc0 + u;
          bool inb = (kr < Kd) && (cc < N);
          float v = Wt[inb ? ((size_t)kr * N + cc) : 0];
          lwh[(sc0 + u) * LWS + skrow] = (_Float16)(inb ? v : 0.0f);
        }
      }
    }
    __syncthreads();

    v16h av = gemm_load_a(arow, k0, khalf, Kd, mvalid, fullM && fullK);
    v16h bv = gemm_load_b(lwh + nlo * LWS, khalf);
    acc = __builtin_amdgcn_wmma_f32_16x16x32_f16(false, av, false, bv,
                                                 (short)0, acc, false, false);
  }

  float bb = 0.0f;
  if (bias) bb = bias[n < N ? n : 0];
#pragma unroll
  for (int r = 0; r < 8; ++r) {
    int mo = tileM * 16 + r + ((lane >> 4) << 3);
    if (mo < M && n < N) Cm[(size_t)mo * N + n] = acc[r] + bb;
  }
}

// Specialized WMMA GEMM for N % 32 == 0: each wave computes a 16x32 output
// strip -> one A fetch feeds TWO wmma per K-chunk (halves A traffic).
__global__ void __launch_bounds__(128) k_gemm2(
    const float* __restrict__ A, const float* __restrict__ Wt,
    const float* __restrict__ bias, float* __restrict__ Cm,
    int M, int Kd, int N) {
  __shared__ _Float16 lwh[32 * LWS];
  const int lane  = threadIdx.x;
  const int wv    = threadIdx.y;
  const int tid   = wv * 32 + lane;
  const int tileM = blockIdx.x * 4 + wv;
  const int mrow  = lane & 15;
  const int nlo   = lane & 15;
  const int khalf = (lane >> 4) << 3;   // 0 or 8
  const int n0    = blockIdx.y * 32;
  const int m     = tileM * 16 + mrow;
  const bool mvalid = (m < M);
  const bool fullM  = (tileM * 16 + 16 <= M);
  const float* arow = A + (size_t)(mvalid ? m : 0) * Kd;

  // staging: thread tid loads 8 consecutive cols of one W row (32x32 chunk)
  const int se    = tid * 8;
  const int skrow = se >> 5;        // 0..31
  const int sc0   = se & 31;        // {0,8,16,24}

  v8f acc0 = {}, acc1 = {};
  for (int k0 = 0; k0 < Kd; k0 += 32) {
    const bool fullK = (k0 + 32 <= Kd);
    if (fullM && (k0 + 64 <= Kd))
      __builtin_prefetch(arow + k0 + 32, 0, 1);

    __syncthreads();
    {
      int kr = k0 + skrow;
      if (fullK) {
        const float* wr = &Wt[(size_t)kr * N + n0 + sc0];
        const float4 w0 = *(const float4*)(wr);
        const float4 w1 = *(const float4*)(wr + 4);
        lwh[(sc0 + 0) * LWS + skrow] = (_Float16)w0.x;
        lwh[(sc0 + 1) * LWS + skrow] = (_Float16)w0.y;
        lwh[(sc0 + 2) * LWS + skrow] = (_Float16)w0.z;
        lwh[(sc0 + 3) * LWS + skrow] = (_Float16)w0.w;
        lwh[(sc0 + 4) * LWS + skrow] = (_Float16)w1.x;
        lwh[(sc0 + 5) * LWS + skrow] = (_Float16)w1.y;
        lwh[(sc0 + 6) * LWS + skrow] = (_Float16)w1.z;
        lwh[(sc0 + 7) * LWS + skrow] = (_Float16)w1.w;
      } else {
#pragma unroll
        for (int u = 0; u < 8; ++u) {
          bool inb = (kr < Kd);
          float v = Wt[inb ? ((size_t)kr * N + n0 + sc0 + u) : 0];
          lwh[(sc0 + u) * LWS + skrow] = (_Float16)(inb ? v : 0.0f);
        }
      }
    }
    __syncthreads();

    v16h av  = gemm_load_a(arow, k0, khalf, Kd, mvalid, fullM && fullK);
    v16h bv0 = gemm_load_b(lwh + nlo * LWS, khalf);
    v16h bv1 = gemm_load_b(lwh + (16 + nlo) * LWS, khalf);
    acc0 = __builtin_amdgcn_wmma_f32_16x16x32_f16(false, av, false, bv0,
                                                  (short)0, acc0, false, false);
    acc1 = __builtin_amdgcn_wmma_f32_16x16x32_f16(false, av, false, bv1,
                                                  (short)0, acc1, false, false);
  }

  float bb0 = bias ? bias[n0 + nlo] : 0.0f;
  float bb1 = bias ? bias[n0 + 16 + nlo] : 0.0f;
#pragma unroll
  for (int r = 0; r < 8; ++r) {
    int mo = tileM * 16 + r + ((lane >> 4) << 3);
    if (mo < M) {
      Cm[(size_t)mo * N + n0 + nlo]      = acc0[r] + bb0;
      Cm[(size_t)mo * N + n0 + 16 + nlo] = acc1[r] + bb1;
    }
  }
}

// BN statistics: per-channel sum/sumsq, LDS reduction then global atomics.
__global__ void k_bnstats(const float* __restrict__ X, size_t Mrows, int C,
                          float* __restrict__ stats) {
  __shared__ float s[256];
  int t = threadIdx.x;
  for (int i = t; i < 2 * C; i += blockDim.x) s[i] = 0.0f;
  __syncthreads();
  size_t total = Mrows * (size_t)C;
  size_t st = (size_t)gridDim.x * blockDim.x;
  for (size_t i = (size_t)blockIdx.x * blockDim.x + t; i < total; i += st) {
    int c = (int)(i % (size_t)C);
    float v = X[i];
    atomicAdd(&s[c], v);          // ds_add_f32
    atomicAdd(&s[C + c], v * v);
  }
  __syncthreads();
  for (int i = t; i < 2 * C; i += blockDim.x) atomicAdd(&stats[i], s[i]);
}

// X = leaky(g*(X-mu)*rsqrt(var+1e-5)+be), in place.
__global__ void k_bnleaky(float* __restrict__ X, size_t Mrows, int C,
                          const float* __restrict__ stats,
                          const float* __restrict__ g, const float* __restrict__ be) {
  size_t total = Mrows * (size_t)C;
  size_t i = (size_t)blockIdx.x * blockDim.x + threadIdx.x;
  if (i >= total) return;
  int c = (int)(i % (size_t)C);
  float inv = 1.0f / (float)Mrows;
  float mu  = stats[c] * inv;
  float var = stats[C + c] * inv - mu * mu;
  float y = g[c] * (X[i] - mu) * rsqrtf(var + 1e-5f) + be[c];
  X[i] = y >= 0.0f ? y : 0.2f * y;
}

__global__ void k_leaky(float* __restrict__ X, size_t n) {
  size_t i = (size_t)blockIdx.x * blockDim.x + threadIdx.x;
  if (i >= n) return;
  float y = X[i];
  X[i] = y >= 0.0f ? y : 0.2f * y;
}

// out[row,c] = max_k msg[(row*16+k),c]
__global__ void k_rowmax(const float* __restrict__ msg, float* __restrict__ out,
                         size_t rows, int C) {
  size_t tot = rows * (size_t)C;
  size_t i = (size_t)blockIdx.x * blockDim.x + threadIdx.x;
  if (i >= tot) return;
  size_t r = i / C; int c = (int)(i % C);
  const float* p = msg + r * (size_t)KNN * C + c;
  float mx = -3.0e38f;
  for (int k = 0; k < KNN; ++k) mx = fmaxf(mx, p[(size_t)k * C]);
  out[i] = mx;
}

// o[e] = pos_s[bm] - pos[b, nbr[e]]
__global__ void k_posdiff(const float* __restrict__ pos_s, const float* __restrict__ pos,
                          const int* __restrict__ nbr, float* __restrict__ o,
                          int B, int N, int M) {
  size_t E = (size_t)B * M * KNN;
  size_t e = (size_t)blockIdx.x * blockDim.x + threadIdx.x;
  if (e >= E) return;
  size_t bm = e >> 4;
  int b = (int)(bm / M);
  int j = nbr[e];
  const float* pi = pos_s + bm * 3;
  const float* pj = pos + ((size_t)b * N + j) * 3;
  o[e * 3 + 0] = pi[0] - pj[0];
  o[e * 3 + 1] = pi[1] - pj[1];
  o[e * 3 + 2] = pi[2] - pj[2];
}

// h[e,c] = x_s[bm,c] - x[b,nbr[e],c] + delta[e,c]
__global__ void k_h(const float* __restrict__ xs, const float* __restrict__ x,
                    const int* __restrict__ nbr, const float* __restrict__ delta,
                    float* __restrict__ h, int B, int N, int M, int C) {
  size_t tot = (size_t)B * M * KNN * C;
  size_t i = (size_t)blockIdx.x * blockDim.x + threadIdx.x;
  if (i >= tot) return;
  size_t e = i / C; int c = (int)(i % C);
  size_t bm = e >> 4;
  int b = (int)(bm / M);
  int j = nbr[e];
  h[i] = xs[bm * (size_t)C + c] - x[((size_t)b * N + j) * C + c] + delta[i];
}

// softmax over the k (=16) axis, in place; data layout [(row*16+k)*C + c]
__global__ void k_softmax_k(float* __restrict__ X, size_t rows, int C) {
  size_t tot = rows * (size_t)C;
  size_t i = (size_t)blockIdx.x * blockDim.x + threadIdx.x;
  if (i >= tot) return;
  size_t row = i / C; int c = (int)(i % C);
  float* p = X + row * (size_t)KNN * C + c;
  float mx = -3.0e38f;
  for (int k = 0; k < KNN; ++k) mx = fmaxf(mx, p[(size_t)k * C]);
  float s = 0.0f;
  for (int k = 0; k < KNN; ++k) { float e = expf(p[(size_t)k * C] - mx); p[(size_t)k * C] = e; s += e; }
  float r = 1.0f / s;
  for (int k = 0; k < KNN; ++k) p[(size_t)k * C] *= r;
}

__global__ void k_wsum_pos(const float* __restrict__ pw, const float* __restrict__ pos,
                           const int* __restrict__ nbr, float* __restrict__ np,
                           int B, int N, int M) {
  int tot = B * M * 3;
  int i = blockIdx.x * blockDim.x + threadIdx.x;
  if (i >= tot) return;
  int bm = i / 3, d = i % 3, b = bm / M;
  float s = 0.0f;
  for (int k = 0; k < KNN; ++k) {
    size_t e = (size_t)bm * KNN + k;
    int j = nbr[e];
    s += pw[e * 3 + d] * pos[((size_t)b * N + j) * 3 + d];
  }
  np[i] = s;
}

__global__ void k_wsum_feat(const float* __restrict__ fw, const float* __restrict__ x,
                            const int* __restrict__ nbr, const float* __restrict__ delta,
                            float* __restrict__ nf, int B, int N, int M, int C) {
  size_t tot = (size_t)B * M * C;
  size_t i = (size_t)blockIdx.x * blockDim.x + threadIdx.x;
  if (i >= tot) return;
  size_t bm = i / C; int c = (int)(i % C);
  int b = (int)(bm / M);
  float s = 0.0f;
  for (int k = 0; k < KNN; ++k) {
    size_t e = bm * KNN + k;
    int j = nbr[e];
    s += fw[e * (size_t)C + c] * (x[((size_t)b * N + j) * C + c] + delta[e * (size_t)C + c]);
  }
  nf[i] = s;
}

// gf[b] = [max_n x3[b,n,:], mean_n x3[b,n,:]]
__global__ void k_globalpool(const float* __restrict__ x, float* __restrict__ gf,
                             int B, int Nn, int C) {
  int i = blockIdx.x * blockDim.x + threadIdx.x;
  if (i >= B * C) return;
  int b = i / C, c = i % C;
  const float* p = x + (size_t)b * Nn * C + c;
  float mx = -3.0e38f, s = 0.0f;
  for (int n = 0; n < Nn; ++n) { float v = p[(size_t)n * C]; mx = fmaxf(mx, v); s += v; }
  gf[(size_t)b * 2 * C + c]     = mx;
  gf[(size_t)b * 2 * C + C + c] = s / (float)Nn;
}

// outputs: coarse[1024,3], feat_max[8,256], batch_out[1024] (as float)
__global__ void k_final(const float* __restrict__ fc3, const float* __restrict__ gf,
                        float* __restrict__ out) {
  int i = blockIdx.x * blockDim.x + threadIdx.x;
  const int NC = 128;
  if (i < BB * NC * 3) {
    int b = i / (NC * 3), rem = i % (NC * 3), j = rem / 3, d = rem % 3;
    out[i] = fc3[b * 384 + d * NC + j];     // (B,3,128) -> (B,128,3)
  } else if (i < BB * NC * 3 + BB * 256) {
    int t = i - BB * NC * 3;
    int b = t / 256, c = t % 256;
    out[i] = gf[b * 512 + c];               // max half of gf
  } else if (i < BB * NC * 3 + BB * 256 + BB * NC) {
    int t = i - (BB * NC * 3 + BB * 256);
    out[i] = (float)(t / NC);               // batch index
  }
}

// ---------------------------------------------------------------------------
// Host side
// ---------------------------------------------------------------------------

struct MLPP { const float *W1, *b1, *g, *be, *W2, *b2; };

static inline unsigned gdiv(size_t n, unsigned b) { return (unsigned)((n + b - 1) / b); }

static void gemm(hipStream_t s, const float* A, const float* W, const float* b,
                 float* C, int M, int Kd, int N) {
  int tm = (M + 15) / 16;
  if ((N & 31) == 0) {
    dim3 blk(32, 4, 1), grd((unsigned)((tm + 3) / 4), (unsigned)(N / 32), 1);
    k_gemm2<<<grd, blk, 0, s>>>(A, W, b, C, M, Kd, N);
  } else {
    int tn = (N + 15) / 16;
    dim3 blk(32, 4, 1), grd((unsigned)((tm + 3) / 4), (unsigned)tn, 1);
    k_gemm<<<grd, blk, 0, s>>>(A, W, b, C, M, Kd, N);
  }
}

static void bn_leaky(hipStream_t s, float* X, size_t Mrows, int C,
                     const float* g, const float* be, float* stats) {
  k_zero<<<4, 256, 0, s>>>(stats, 1024);
  size_t total = Mrows * (size_t)C;
  size_t nb = (total + 255) / 256;
  unsigned sb = nb > 4096 ? 4096u : (unsigned)nb;
  k_bnstats<<<sb, 256, 0, s>>>(X, Mrows, C, stats);
  k_bnleaky<<<gdiv(total, 256), 256, 0, s>>>(X, Mrows, C, stats, g, be);
}

extern "C" void kernel_launch(void* const* d_in, const int* in_sizes, int n_in,
                              void* d_out, int out_size, void* d_ws, size_t ws_size,
                              hipStream_t stream) {
  (void)n_in; (void)out_size; (void)ws_size;
  auto F = [&](int i) { return (const float*)d_in[i]; };

  const float* pos;
  MLPP g1, g2, g3, p1pm, p1fa, p1pa, p2pm, p2fa, p2pa;
  const float *fcW1, *fcb1, *fcW2, *fcb2, *fcW3, *fcb3;

  if (in_sizes[0] == BB * N0 * 3) {
    // dict insertion order: pos, g1, g2, g3, pool1{pos_mlp,feat_attn,pos_attn},
    // pool2{...}, fc ; each mlp2 = W1,b1,g,be,W2,b2
    auto M = [&](int base) { MLPP m{F(base), F(base + 1), F(base + 2), F(base + 3), F(base + 4), F(base + 5)}; return m; };
    pos = F(0);
    g1 = M(1); g2 = M(7); g3 = M(13);
    p1pm = M(19); p1fa = M(25); p1pa = M(31);
    p2pm = M(37); p2fa = M(43); p2pa = M(49);
    fcW1 = F(55); fcb1 = F(56); fcW2 = F(57); fcb2 = F(58); fcW3 = F(59); fcb3 = F(60);
  } else {
    // jax tree-flatten (sorted keys): params{fc,g1,g2,g3,pool1,pool2}, pos
    // fc: W1,W2,W3,b1,b2,b3 ; mlp2: W1,W2,b1,b2,be,g ; pool: feat_attn,pos_attn,pos_mlp
    auto M = [&](int base) { MLPP m{F(base), F(base + 2), F(base + 5), F(base + 4), F(base + 1), F(base + 3)}; return m; };
    fcW1 = F(0); fcW2 = F(1); fcW3 = F(2); fcb1 = F(3); fcb2 = F(4); fcb3 = F(5);
    g1 = M(6); g2 = M(12); g3 = M(18);
    p1fa = M(24); p1pa = M(30); p1pm = M(36);
    p2fa = M(42); p2pa = M(48); p2pm = M(54);
    pos = F(60);
  }

  float* ws = (float*)d_ws;
  float* stats = ws + O_STATS;
  int* idx1 = (int*)(ws + O_IDX1);
  int* fps1 = (int*)(ws + O_FPS1);
  int* nbr1 = (int*)(ws + O_NBR1);
  int* idx2 = (int*)(ws + O_IDX2);
  int* fps2 = (int*)(ws + O_FPS2);
  int* nbr2 = (int*)(ws + O_NBR2);
  int* idx3 = (int*)(ws + O_IDX3);

  // ---------------- g1: dynamic edge conv on pos (3 -> 6 -> 32 -> 64) --------
  k_knn<3><<<gdiv((size_t)BB * N0, 128), 128, 0, stream>>>(pos, pos, BB, N0, N0, idx1);
  k_edgefeat<<<gdiv(E1, 256), 256, 0, stream>>>(pos, idx1, ws + A_EF1, BB, N0, 3);
  gemm(stream, ws + A_EF1, g1.W1, g1.b1, ws + A_HPRE1, (int)E1, 6, 32);
  bn_leaky(stream, ws + A_HPRE1, E1, 32, g1.g, g1.be, stats);
  gemm(stream, ws + A_HPRE1, g1.W2, g1.b2, ws + A_MSG1, (int)E1, 32, 64);
  k_rowmax<<<gdiv((size_t)BB * N0 * 64, 256), 256, 0, stream>>>(ws + A_MSG1, ws + O_X1, (size_t)BB * N0, 64);

  // ---------------- pool1: 2048 -> 512, C=64 ---------------------------------
  k_fps<<<BB, 256, 0, stream>>>(pos, N0, M1, fps1);
  k_gather<<<gdiv((size_t)BB * M1 * 3, 256), 256, 0, stream>>>(pos, fps1, ws + O_POS1S, BB, N0, M1, 3);
  k_gather<<<gdiv((size_t)BB * M1 * 64, 256), 256, 0, stream>>>(ws + O_X1, fps1, ws + O_X1S, BB, N0, M1, 64);
  k_knn<3><<<gdiv((size_t)BB * M1, 128), 128, 0, stream>>>(ws + O_POS1S, pos, BB, M1, N0, nbr1);
  k_posdiff<<<gdiv(EP1, 256), 256, 0, stream>>>(ws + O_POS1S, pos, nbr1, ws + A_PD1, BB, N0, M1);
  // pos_mlp: 3 -> 16 -> 64
  gemm(stream, ws + A_PD1, p1pm.W1, p1pm.b1, ws + A_DH1, (int)EP1, 3, 16);
  bn_leaky(stream, ws + A_DH1, EP1, 16, p1pm.g, p1pm.be, stats);
  gemm(stream, ws + A_DH1, p1pm.W2, p1pm.b2, ws + A_DL1, (int)EP1, 16, 64);
  // h = x_i - x_j + delta
  k_h<<<gdiv(EP1 * 64, 256), 256, 0, stream>>>(ws + O_X1S, ws + O_X1, nbr1, ws + A_DL1, ws + A_H1, BB, N0, M1, 64);
  // pos_attn: 64 -> 16 -> 3, softmax over k
  gemm(stream, ws + A_H1, p1pa.W1, p1pa.b1, ws + A_AH1, (int)EP1, 64, 16);
  bn_leaky(stream, ws + A_AH1, EP1, 16, p1pa.g, p1pa.be, stats);
  gemm(stream, ws + A_AH1, p1pa.W2, p1pa.b2, ws + A_PW1, (int)EP1, 16, 3);
  k_softmax_k<<<gdiv((size_t)BB * M1 * 3, 256), 256, 0, stream>>>(ws + A_PW1, (size_t)BB * M1, 3);
  // feat_attn: 64 -> 16 -> 64, softmax over k
  gemm(stream, ws + A_H1, p1fa.W1, p1fa.b1, ws + A_AH1, (int)EP1, 64, 16);
  bn_leaky(stream, ws + A_AH1, EP1, 16, p1fa.g, p1fa.be, stats);
  gemm(stream, ws + A_AH1, p1fa.W2, p1fa.b2, ws + A_FW1, (int)EP1, 16, 64);
  k_softmax_k<<<gdiv((size_t)BB * M1 * 64, 256), 256, 0, stream>>>(ws + A_FW1, (size_t)BB * M1, 64);
  // weighted sums
  k_wsum_pos<<<gdiv((size_t)BB * M1 * 3, 256), 256, 0, stream>>>(ws + A_PW1, pos, nbr1, ws + O_NP1, BB, N0, M1);
  k_wsum_feat<<<gdiv((size_t)BB * M1 * 64, 256), 256, 0, stream>>>(ws + A_FW1, ws + O_X1, nbr1, ws + A_DL1, ws + O_NF1, BB, N0, M1, 64);

  // ---------------- g2: edge conv on features (64 -> 128 -> 64 -> 128) -------
  k_knn<64><<<gdiv((size_t)BB * M1, 128), 128, 0, stream>>>(ws + O_NF1, ws + O_NF1, BB, M1, M1, idx2);
  k_edgefeat<<<gdiv(EP1, 256), 256, 0, stream>>>(ws + O_NF1, idx2, ws + A_EF2, BB, M1, 64);
  gemm(stream, ws + A_EF2, g2.W1, g2.b1, ws + A_HP2, (int)EP1, 128, 64);
  bn_leaky(stream, ws + A_HP2, EP1, 64, g2.g, g2.be, stats);
  gemm(stream, ws + A_HP2, g2.W2, g2.b2, ws + A_MSG2, (int)EP1, 64, 128);
  k_rowmax<<<gdiv((size_t)BB * M1 * 128, 256), 256, 0, stream>>>(ws + A_MSG2, ws + O_X2, (size_t)BB * M1, 128);

  // ---------------- pool2: 512 -> 256, C=128 ---------------------------------
  k_fps<<<BB, 256, 0, stream>>>(ws + O_NP1, M1, M2, fps2);
  k_gather<<<gdiv((size_t)BB * M2 * 3, 256), 256, 0, stream>>>(ws + O_NP1, fps2, ws + O_POS2S, BB, M1, M2, 3);
  k_gather<<<gdiv((size_t)BB * M2 * 128, 256), 256, 0, stream>>>(ws + O_X2, fps2, ws + O_X2S, BB, M1, M2, 128);
  k_knn<3><<<gdiv((size_t)BB * M2, 128), 128, 0, stream>>>(ws + O_POS2S, ws + O_NP1, BB, M2, M1, nbr2);
  k_posdiff<<<gdiv(EP2, 256), 256, 0, stream>>>(ws + O_POS2S, ws + O_NP1, nbr2, ws + A_PD2, BB, M1, M2);
  gemm(stream, ws + A_PD2, p2pm.W1, p2pm.b1, ws + A_DH2, (int)EP2, 3, 16);
  bn_leaky(stream, ws + A_DH2, EP2, 16, p2pm.g, p2pm.be, stats);
  gemm(stream, ws + A_DH2, p2pm.W2, p2pm.b2, ws + A_DL2, (int)EP2, 16, 128);
  k_h<<<gdiv(EP2 * 128, 256), 256, 0, stream>>>(ws + O_X2S, ws + O_X2, nbr2, ws + A_DL2, ws + A_H2, BB, M1, M2, 128);
  gemm(stream, ws + A_H2, p2pa.W1, p2pa.b1, ws + A_AH2, (int)EP2, 128, 16);
  bn_leaky(stream, ws + A_AH2, EP2, 16, p2pa.g, p2pa.be, stats);
  gemm(stream, ws + A_AH2, p2pa.W2, p2pa.b2, ws + A_PW2, (int)EP2, 16, 3);
  k_softmax_k<<<gdiv((size_t)BB * M2 * 3, 256), 256, 0, stream>>>(ws + A_PW2, (size_t)BB * M2, 3);
  gemm(stream, ws + A_H2, p2fa.W1, p2fa.b1, ws + A_AH2, (int)EP2, 128, 16);
  bn_leaky(stream, ws + A_AH2, EP2, 16, p2fa.g, p2fa.be, stats);
  gemm(stream, ws + A_AH2, p2fa.W2, p2fa.b2, ws + A_FW2, (int)EP2, 16, 128);
  k_softmax_k<<<gdiv((size_t)BB * M2 * 128, 256), 256, 0, stream>>>(ws + A_FW2, (size_t)BB * M2, 128);
  k_wsum_pos<<<gdiv((size_t)BB * M2 * 3, 256), 256, 0, stream>>>(ws + A_PW2, ws + O_NP1, nbr2, ws + O_NP2, BB, M1, M2);
  k_wsum_feat<<<gdiv((size_t)BB * M2 * 128, 256), 256, 0, stream>>>(ws + A_FW2, ws + O_X2, nbr2, ws + A_DL2, ws + O_NF2, BB, M1, M2, 128);

  // ---------------- g3: edge conv (128 -> 256 -> 128 -> 256) -----------------
  k_knn<128><<<gdiv((size_t)BB * M2, 128), 128, 0, stream>>>(ws + O_NF2, ws + O_NF2, BB, M2, M2, idx3);
  k_edgefeat<<<gdiv(EP2, 256), 256, 0, stream>>>(ws + O_NF2, idx3, ws + A_EF3, BB, M2, 128);
  gemm(stream, ws + A_EF3, g3.W1, g3.b1, ws + A_HP3, (int)EP2, 256, 128);
  bn_leaky(stream, ws + A_HP3, EP2, 128, g3.g, g3.be, stats);
  gemm(stream, ws + A_HP3, g3.W2, g3.b2, ws + A_MSG3, (int)EP2, 128, 256);
  k_rowmax<<<gdiv((size_t)BB * M2 * 256, 256), 256, 0, stream>>>(ws + A_MSG3, ws + O_X3, (size_t)BB * M2, 256);

  // ---------------- head: global pool + FC -----------------------------------
  k_globalpool<<<gdiv((size_t)BB * 256, 256), 256, 0, stream>>>(ws + O_X3, ws + O_GF, BB, M2, 256);
  gemm(stream, ws + O_GF, fcW1, fcb1, ws + O_FC1, BB, 512, 512);
  k_leaky<<<gdiv((size_t)BB * 512, 256), 256, 0, stream>>>(ws + O_FC1, (size_t)BB * 512);
  gemm(stream, ws + O_FC1, fcW2, fcb2, ws + O_FC2, BB, 512, 512);
  k_leaky<<<gdiv((size_t)BB * 512, 256), 256, 0, stream>>>(ws + O_FC2, (size_t)BB * 512);
  gemm(stream, ws + O_FC2, fcW3, fcb3, ws + O_FC3, BB, 512, 384);

  k_final<<<gdiv(6144, 256), 256, 0, stream>>>(ws + O_FC3, ws + O_GF, (float*)d_out);
}
